// LinBlock_23021024707079
// MI455X (gfx1250) — matrix operand order repaired and checked by reading, not compiled, not run on hardware
//
#include <hip/hip_runtime.h>
#include <hip/hip_bf16.h>

// ---------------------------------------------------------------------------
// Types for CDNA5 WMMA (wave32): v_wmma_f32_16x16x32_bf16
// ---------------------------------------------------------------------------
typedef __bf16 bf16_t;
typedef bf16_t v16bf  __attribute__((ext_vector_type(16)));
typedef bf16_t bf16x8 __attribute__((ext_vector_type(8)));
typedef bf16_t bf16x4 __attribute__((ext_vector_type(4)));
typedef float  v8f    __attribute__((ext_vector_type(8)));

static __device__ __forceinline__ v16bf frag_cat(bf16x8 lo, bf16x8 hi) {
    v16bf r;
#pragma unroll
    for (int i = 0; i < 8; ++i) { r[i] = lo[i]; r[i + 8] = hi[i]; }
    return r;
}

// A-fragment (16x32, row-major source): lane holds row = `row`, two contiguous
// 16B chunks at k = k0 + half*8 and k0 + 16 + half*8  (ISA 7.12.2).
static __device__ __forceinline__ v16bf load_a_frag(const bf16_t* base, int ld,
                                                    int row, int k0, int lane) {
    const int h = lane >> 4;
    const bf16_t* p = base + (size_t)row * ld + k0 + h * 8;
    return frag_cat(*(const bf16x8*)p, *(const bf16x8*)(p + 16));
}

// B-fragment (32x16) from B^T-form storage (row n holds contiguous K):
// lane holds col = `col`, two 16B chunks at k = k0 + half*8 and k0 + 16 + half*8.
static __device__ __forceinline__ v16bf load_b_frag(const bf16_t* base, int ld,
                                                    int col, int k0, int lane) {
    const int h = lane >> 4;
    const bf16_t* p = base + (size_t)col * ld + k0 + h * 8;
    return frag_cat(*(const bf16x8*)p, *(const bf16x8*)(p + 16));
}

// Async global->LDS 16B move (gfx1250): the immediate offset is added to both the LDS and the global address.
// Tracked by ASYNCcnt. ldsoff is a byte offset in the wave's LDS allocation.
static __device__ __forceinline__ void async_ld_b128(unsigned ldsoff,
                                                     unsigned long long gaddr,
                                                     int ioff) {
    switch (ioff) {
    case 0:  asm volatile("global_load_async_to_lds_b128 %0, %1, off"
                          :: "v"(ldsoff), "v"(gaddr) : "memory"); break;
    case 16: asm volatile("global_load_async_to_lds_b128 %0, %1, off offset:16"
                          :: "v"(ldsoff), "v"(gaddr) : "memory"); break;
    case 32: asm volatile("global_load_async_to_lds_b128 %0, %1, off offset:32"
                          :: "v"(ldsoff), "v"(gaddr) : "memory"); break;
    default: asm volatile("global_load_async_to_lds_b128 %0, %1, off offset:48"
                          :: "v"(ldsoff), "v"(gaddr) : "memory"); break;
    }
}

static __device__ __forceinline__ void wait_async0() {
    asm volatile("s_wait_asynccnt 0x0" ::: "memory");
}

// ---------------------------------------------------------------------------
// Elementwise f32 -> bf16 cast, x4 vectorized (all sizes are multiples of 4)
// ---------------------------------------------------------------------------
__global__ void cast_f32_bf16(const float* __restrict__ in,
                              bf16_t* __restrict__ out, size_t n) {
    const size_t i = (size_t)blockIdx.x * blockDim.x + threadIdx.x;
    if (i * 4 < n) {
        const float4 v = ((const float4*)in)[i];
        bf16x4 o;
        o[0] = (bf16_t)v.x; o[1] = (bf16_t)v.y;
        o[2] = (bf16_t)v.z; o[3] = (bf16_t)v.w;
        ((bf16x4*)out)[i] = o;
    }
}

// ---------------------------------------------------------------------------
// Tiled transpose + cast: in[rows][cols] f32 -> out[cols][rows] bf16
// ---------------------------------------------------------------------------
__global__ __launch_bounds__(256) void transpose_cast(const float* __restrict__ in,
                                                      bf16_t* __restrict__ out,
                                                      int rows, int cols) {
    __shared__ float tile[32][33];
    const int c0 = blockIdx.x * 32, r0 = blockIdx.y * 32;
    const int tx = threadIdx.x & 31, ty = threadIdx.x >> 5;  // 32x8
#pragma unroll
    for (int i = ty; i < 32; i += 8)
        tile[i][tx] = in[(size_t)(r0 + i) * cols + c0 + tx];
    __syncthreads();
#pragma unroll
    for (int i = ty; i < 32; i += 8)
        out[(size_t)(c0 + i) * rows + r0 + tx] = (bf16_t)tile[tx][i];
}

// ---------------------------------------------------------------------------
// WMMA GEMM: C[M,N] = (A[M,K] * B[N,K]^T + bias[N]) * alpha
// A,B bf16 row-major. Block tile 128x64, 256 threads (8 waves, 4x2 grid),
// each wave a 32x32 tile (2x2 wmma accumulators), K stepped by 64
// (8 v_wmma_f32_16x16x32_bf16 per LDS stage).
// Staging uses gfx1250 GLOBAL_LOAD_ASYNC_TO_LDS_B128 (ASYNCcnt) with
// double-buffered dynamic LDS: next tile's DMA overlaps current WMMAs.
// LDS layout (dynamic segment, base offset 0):
//   buf b: A at b*13824, pitch 72 elts (144B); B at b*13824 + 9216.
// Requires M%128==0, N%64==0, K%64==0 (true for all uses here).
// ---------------------------------------------------------------------------
extern __shared__ bf16_t g_smem[];

template <bool HAS_BIAS, bool OUT_F32>
__global__ __launch_bounds__(256) void gemm_bt_wmma(
    const bf16_t* __restrict__ A, const bf16_t* __restrict__ B,
    void* __restrict__ Cout, const float* __restrict__ bias,
    int M, int N, int K, float alpha) {
    const int tid  = threadIdx.x;
    const int lane = tid & 31;
    const int wave = tid >> 5;
    const int ln   = lane & 15;
    const int bm = blockIdx.x * 128;
    const int bn = blockIdx.y * 64;
    const int mwave = (wave >> 1) * 32;  // 0,32,64,96
    const int nwave = (wave & 1) * 32;   // 0,32

    // staging assignment (per thread: A 4x16B, B 2x16B; contiguous both sides)
    const int arow = tid >> 1;           // 0..127
    const int acol = (tid & 1) * 32;     // 0 or 32
    const int brow = tid >> 2;           // 0..63
    const int bcol = (tid & 3) * 16;     // 0,16,32,48

    const bf16_t* pa_base = A + (size_t)(bm + arow) * K + acol;
    const bf16_t* pb_base = B + (size_t)(bn + brow) * K + bcol;
    const unsigned ldsA0 = (unsigned)((arow * 72 + acol) * 2);
    const unsigned ldsB0 = (unsigned)((9216 + brow * 72 + bcol) * 2);

    v8f acc[2][2] = {};

    auto stage_async = [&](int k0, int buf) {
        const unsigned boff = (unsigned)(buf * 13824 * 2);
        const unsigned long long gA = (unsigned long long)(uintptr_t)(pa_base + k0);
#pragma unroll
        for (int c = 0; c < 4; ++c) async_ld_b128(ldsA0 + boff, gA, c * 16);
        const unsigned long long gB = (unsigned long long)(uintptr_t)(pb_base + k0);
#pragma unroll
        for (int c = 0; c < 2; ++c) async_ld_b128(ldsB0 + boff, gB, c * 16);
    };

    stage_async(0, 0);
    int cur = 0;
    for (int k0 = 0; k0 < K; k0 += 64) {
        wait_async0();      // own DMA for buf[cur] complete
        __syncthreads();    // everyone's DMA complete -> stage visible
        if (k0 + 64 < K) stage_async(k0 + 64, cur ^ 1);  // overlaps WMMAs below

        const bf16_t* sAb = g_smem + cur * 13824;
        const bf16_t* sBb = g_smem + cur * 13824 + 9216;
#pragma unroll
        for (int ks = 0; ks < 64; ks += 32) {
            v16bf af[2], bfr[2];
            af[0]  = load_a_frag(sAb, 72, mwave + ln, ks, lane);
            af[1]  = load_a_frag(sAb, 72, mwave + 16 + ln, ks, lane);
            bfr[0] = load_b_frag(sBb, 72, nwave + ln, ks, lane);
            bfr[1] = load_b_frag(sBb, 72, nwave + 16 + ln, ks, lane);
#pragma unroll
            for (int i = 0; i < 2; ++i)
#pragma unroll
                for (int j = 0; j < 2; ++j)
                    acc[i][j] = __builtin_amdgcn_wmma_f32_16x16x32_bf16(
                        false, af[i], false, bfr[j], (short)0, acc[i][j],
                        false, false);
        }
        cur ^= 1;
    }

    // Branch-free epilogue. C-frag: vgpr r -> row = r + 8*(lane>=16), col = ln.
    const int half = lane >> 4;
#pragma unroll
    for (int i = 0; i < 2; ++i) {
        const int gm0 = bm + mwave + i * 16 + 8 * half;
#pragma unroll
        for (int j = 0; j < 2; ++j) {
            const int gn = bn + nwave + j * 16 + ln;
            const float bv = HAS_BIAS ? bias[gn] : 0.0f;
            if constexpr (OUT_F32) {
                float* p = (float*)Cout + (size_t)gm0 * N + gn;
#pragma unroll
                for (int r = 0; r < 8; ++r)
                    p[(size_t)r * N] = (acc[i][j][r] + bv) * alpha;
            } else {
                bf16_t* p = (bf16_t*)Cout + (size_t)gm0 * N + gn;
#pragma unroll
                for (int r = 0; r < 8; ++r)
                    p[(size_t)r * N] = (bf16_t)((acc[i][j][r] + bv) * alpha);
            }
        }
    }
}

// ---------------------------------------------------------------------------
// Flash attention over compressed keys.
// Q,K,V,O bf16 laid out [L, B*E]; head (b,h) lives at column b*E + h*64.
// Grid: (T/64, B*H). Block: 128 threads = 4 waves; each wave owns 16 q rows.
// Per 32-key chunk: K,V staged in LDS once per block, S = Q·K^T via 4 wmma,
// online softmax (stage-batched shfl_xor max; row-sum via P·ones WMMA),
// P->LDS relayout, O += P·V via 4 wmma.  9 WMMAs / chunk / wave.
// ---------------------------------------------------------------------------
__global__ __launch_bounds__(128) void flash_attn_wmma(
    const bf16_t* __restrict__ Q, const bf16_t* __restrict__ Km,
    const bf16_t* __restrict__ Vm, bf16_t* __restrict__ O,
    int T, int CK, int Bsz, int E) {
    __shared__ bf16_t sK[32][72];   // K chunk,       [key][hd]      (pitch 144B)
    __shared__ bf16_t sVT[64][40];  // V transposed,  [hd][key]      (pitch 80B)
    __shared__ bf16_t sP[64][40];   // probabilities, [q-row][key]   (pitch 80B)

    const int tid  = threadIdx.x;
    const int lane = tid & 31;
    const int wave = tid >> 5;
    const int half = lane >> 4;
    const int ln   = lane & 15;

    const int qbase = blockIdx.x * 64;
    const int bh = blockIdx.y;          // b*8 + h
    const int b = bh >> 3, h = bh & 7;
    const int rowstride = Bsz * E;      // 2048
    const size_t headoff = (size_t)b * E + h * 64;

    // Persistent Q fragments for this wave's 16 rows (hd=64 -> 2 k-steps).
    const int qrow = qbase + wave * 16 + ln;
    const bf16_t* qptr = Q + (size_t)qrow * rowstride + headoff;
    v16bf qa[2];
#pragma unroll
    for (int ks = 0; ks < 2; ++ks) {
        const bf16_t* p = qptr + ks * 32 + half * 8;
        qa[ks] = frag_cat(*(const bf16x8*)p, *(const bf16x8*)(p + 16));
    }

    // all-ones B fragment: row-sums of P via one extra WMMA
    v16bf ones;
#pragma unroll
    for (int i = 0; i < 16; ++i) ones[i] = (bf16_t)1.0f;

    v8f o[4] = {};
    float mrun[8], lrun[8];
#pragma unroll
    for (int r = 0; r < 8; ++r) { mrun[r] = -3.0e38f; lrun[r] = 0.0f; }

    for (int kc = 0; kc < CK; kc += 32) {
        __syncthreads();  // previous iteration's sK/sVT reads done

        // Stage K chunk: 32 keys x 64 hd (each thread: 16 contiguous bf16).
        {
            const int key = tid & 31;
            const int cg  = tid >> 5;  // 0..3 -> hd group of 16
            const bf16_t* kp = Km + (size_t)(kc + key) * rowstride + headoff + cg * 16;
            *(bf16x8*)&sK[key][cg * 16]     = *(const bf16x8*)kp;
            *(bf16x8*)&sK[key][cg * 16 + 8] = *(const bf16x8*)(kp + 8);
        }
        // Stage V chunk transposed: 32 keys x 64 hd -> sVT[hd][key].
#pragma unroll
        for (int t = tid; t < 256; t += 128) {
            const int key = t & 31;
            const int cg  = t >> 5;  // hd group of 8
            const bf16_t* vp = Vm + (size_t)(kc + key) * rowstride + headoff + cg * 8;
            bf16x8 vv = *(const bf16x8*)vp;
#pragma unroll
            for (int e = 0; e < 8; ++e) sVT[cg * 8 + e][key] = vv[e];
        }
        __syncthreads();  // sK/sVT staged

        // S = Q · K^T: load all four B-fragments first (batched ds waits).
        v16bf kb[2][2];
#pragma unroll
        for (int nt = 0; nt < 2; ++nt)
#pragma unroll
            for (int ks = 0; ks < 2; ++ks)
                kb[nt][ks] = load_b_frag(&sK[0][0], 72, nt * 16 + ln, ks * 32, lane);
        v8f s[2] = {};
#pragma unroll
        for (int nt = 0; nt < 2; ++nt)
#pragma unroll
            for (int ks = 0; ks < 2; ++ks)
                s[nt] = __builtin_amdgcn_wmma_f32_16x16x32_bf16(
                    false, qa[ks], false, kb[nt][ks], (short)0, s[nt], false, false);

        // Row max: batched butterfly (8 independent bpermutes per stage).
        float mx[8];
#pragma unroll
        for (int r = 0; r < 8; ++r) mx[r] = fmaxf(s[0][r], s[1][r]);
#pragma unroll
        for (int st = 1; st <= 8; st <<= 1)
#pragma unroll
            for (int r = 0; r < 8; ++r)
                mx[r] = fmaxf(mx[r], __shfl_xor(mx[r], st, 32));

        // exp, rescale O, write P (row = r + 8*half).
        float corr[8];
#pragma unroll
        for (int r = 0; r < 8; ++r) {
            const float mnew = fmaxf(mrun[r], mx[r]);
            corr[r] = __expf(mrun[r] - mnew);
            mrun[r] = mnew;
            const float p0 = __expf(s[0][r] - mnew);
            const float p1 = __expf(s[1][r] - mnew);
            const int prow = wave * 16 + r + 8 * half;
            sP[prow][ln]      = (bf16_t)p0;
            sP[prow][16 + ln] = (bf16_t)p1;
#pragma unroll
            for (int j = 0; j < 4; ++j) o[j][r] *= corr[r];
        }

        // P fragment (own wave's rows only; DScnt keeps wave-local order).
        v16bf pa;
        {
            const bf16_t* pp = &sP[wave * 16 + ln][half * 8];
            pa = frag_cat(*(const bf16x8*)pp, *(const bf16x8*)(pp + 16));
        }

        // Row-sums of P via WMMA against all-ones B; layout matches lrun.
        v8f psum = {};
        psum = __builtin_amdgcn_wmma_f32_16x16x32_bf16(
            false, pa, false, ones, (short)0, psum, false, false);
#pragma unroll
        for (int r = 0; r < 8; ++r) lrun[r] = lrun[r] * corr[r] + psum[r];

        // O += P · V: load all four V-fragments first, then 4 WMMAs.
        v16bf vb[4];
#pragma unroll
        for (int nt = 0; nt < 4; ++nt) {
            const bf16_t* vtp = &sVT[nt * 16 + ln][half * 8];
            vb[nt] = frag_cat(*(const bf16x8*)vtp, *(const bf16x8*)(vtp + 16));
        }
#pragma unroll
        for (int nt = 0; nt < 4; ++nt)
            o[nt] = __builtin_amdgcn_wmma_f32_16x16x32_bf16(
                false, pa, false, vb[nt], (short)0, o[nt], false, false);
    }

    // Normalize and store (bf16) to O[t, b*E + h*64 + d].
    float rinv[8];
#pragma unroll
    for (int r = 0; r < 8; ++r) rinv[r] = 1.0f / lrun[r];
    const int trow0 = qbase + wave * 16 + 8 * half;
#pragma unroll
    for (int nt = 0; nt < 4; ++nt) {
        bf16_t* p = O + (size_t)trow0 * rowstride + headoff + nt * 16 + ln;
#pragma unroll
        for (int r = 0; r < 8; ++r)
            p[(size_t)r * rowstride] = (bf16_t)(o[nt][r] * rinv[r]);
    }
}

// ---------------------------------------------------------------------------
// Host orchestration
// ---------------------------------------------------------------------------
extern "C" void kernel_launch(void* const* d_in, const int* in_sizes, int n_in,
                              void* d_out, int out_size, void* d_ws, size_t ws_size,
                              hipStream_t stream) {
    (void)in_sizes; (void)n_in; (void)out_size; (void)ws_size;
    const int T = 4096, Bsz = 4, E = 512, CK = 1024;
    const float scaling = 0.125f;  // (E/H)^-0.5 = 64^-0.5
    const size_t gemm_lds = 2 * 13824 * sizeof(bf16_t);  // 55296 B, double buffer

    const float* query = (const float*)d_in[0];
    const float* Wq = (const float*)d_in[1];  const float* bq = (const float*)d_in[2];
    const float* Wk = (const float*)d_in[3];  const float* bk = (const float*)d_in[4];
    const float* Wv = (const float*)d_in[5];  const float* bv = (const float*)d_in[6];
    const float* Wck = (const float*)d_in[7]; const float* Wcv = (const float*)d_in[8];
    const float* Wo = (const float*)d_in[9];  const float* bo = (const float*)d_in[10];

    const size_t nQ = (size_t)T * Bsz * E;    // 8388608
    const size_t nW = (size_t)E * E;          // 262144
    const size_t nWc = (size_t)CK * T;        // 4194304
    const size_t nKin = (size_t)CK * Bsz * E; // 2097152

    char* ws = (char*)d_ws;
    size_t off = 0;
    auto alloc = [&](size_t elems) { bf16_t* p = (bf16_t*)(ws + off); off += elems * 2; return p; };
    bf16_t* qry_bf = alloc(nQ);
    bf16_t* QT_bf  = alloc(nQ);     // [B*E, T]
    bf16_t* Wq_bf  = alloc(nW);
    bf16_t* Wk_bf  = alloc(nW);
    bf16_t* Wv_bf  = alloc(nW);
    bf16_t* Wo_bf  = alloc(nW);
    bf16_t* Wck_bf = alloc(nWc);
    bf16_t* Wcv_bf = alloc(nWc);
    bf16_t* q_bf   = alloc(nQ);     // scaled q projection
    bf16_t* kin_bf = alloc(nKin);
    bf16_t* vin_bf = alloc(nKin);
    bf16_t* k_bf   = alloc(nKin);
    bf16_t* v_bf   = alloc(nKin);
    bf16_t* att_bf = alloc(nQ);

    auto cast = [&](const float* src, bf16_t* dst, size_t n) {
        cast_f32_bf16<<<dim3((unsigned)(n / 4 + 255) / 256), dim3(256), 0, stream>>>(src, dst, n);
    };
    cast(query, qry_bf, nQ);
    cast(Wq, Wq_bf, nW);  cast(Wk, Wk_bf, nW);
    cast(Wv, Wv_bf, nW);  cast(Wo, Wo_bf, nW);
    cast(Wck, Wck_bf, nWc); cast(Wcv, Wcv_bf, nWc);

    // QT[j, t] = query[t, j]  (rows=T=4096, cols=B*E=2048)
    transpose_cast<<<dim3(2048 / 32, 4096 / 32), dim3(256), 0, stream>>>(
        query, QT_bf, T, Bsz * E);

    // q = (query @ Wq^T + bq) * scaling     [16384, 512] -> bf16
    gemm_bt_wmma<true, false><<<dim3((T * Bsz) / 128, E / 64), dim3(256), gemm_lds, stream>>>(
        qry_bf, Wq_bf, q_bf, bq, T * Bsz, E, E, scaling);

    // k_in = Wck @ query_2d                 [1024, 2048] -> bf16
    gemm_bt_wmma<false, false><<<dim3(CK / 128, (Bsz * E) / 64), dim3(256), gemm_lds, stream>>>(
        Wck_bf, QT_bf, kin_bf, nullptr, CK, Bsz * E, T, 1.0f);
    gemm_bt_wmma<false, false><<<dim3(CK / 128, (Bsz * E) / 64), dim3(256), gemm_lds, stream>>>(
        Wcv_bf, QT_bf, vin_bf, nullptr, CK, Bsz * E, T, 1.0f);

    // k = k_in @ Wk^T + bk ; v = v_in @ Wv^T + bv    ([CK*B, E] view) -> bf16
    gemm_bt_wmma<true, false><<<dim3((CK * Bsz) / 128, E / 64), dim3(256), gemm_lds, stream>>>(
        kin_bf, Wk_bf, k_bf, bk, CK * Bsz, E, E, 1.0f);
    gemm_bt_wmma<true, false><<<dim3((CK * Bsz) / 128, E / 64), dim3(256), gemm_lds, stream>>>(
        vin_bf, Wv_bf, v_bf, bv, CK * Bsz, E, E, 1.0f);

    // attention (flash, per head), output bf16 [T, B*E]
    flash_attn_wmma<<<dim3(T / 64, Bsz * 8), dim3(128), 0, stream>>>(
        q_bf, k_bf, v_bf, att_bf, T, CK, Bsz, E);

    // out = attn_out @ Wo^T + bo  -> fp32 d_out
    gemm_bt_wmma<true, true><<<dim3((T * Bsz) / 128, E / 64), dim3(256), gemm_lds, stream>>>(
        att_bf, Wo_bf, d_out, bo, T * Bsz, E, E, 1.0f);
}
